// Attention_1503238553757
// MI455X (gfx1250) — matrix operand order repaired and checked by reading, not compiled, not run on hardware
//
#include <hip/hip_runtime.h>
#include <hip/hip_bf16.h>

// ---------------------------------------------------------------------------
// Types for CDNA5 WMMA (wave32): v_wmma_f32_16x16x32_bf16
// ---------------------------------------------------------------------------
typedef __attribute__((ext_vector_type(16))) __bf16 v16bf;
typedef __attribute__((ext_vector_type(8)))  float  v8f;

union Frag16 {
    v16bf v;
    unsigned short u[16];
    uint4 q[2];
};

__device__ __forceinline__ unsigned short f2bf(float x) {
    unsigned int u = __float_as_uint(x);
    unsigned int r = u + 0x7FFFu + ((u >> 16) & 1u);   // round-to-nearest-even
    return (unsigned short)(r >> 16);
}

#define WMMA_BF16(A, B, C) \
    __builtin_amdgcn_wmma_f32_16x16x32_bf16(false, (A), false, (B), (short)0, (C), false, false)

// CDNA5 async copy: 16 bytes global -> LDS, tracked by ASYNCcnt
__device__ __forceinline__ void async_copy16(unsigned lds_addr, const void* gptr) {
    unsigned long long ga = (unsigned long long)gptr;
    asm volatile("global_load_async_to_lds_b128 %0, %1, off"
                 :: "v"(lds_addr), "v"(ga) : "memory");
}
__device__ __forceinline__ void wait_async0() {
    asm volatile("s_wait_asynccnt 0" ::: "memory");
}

// Problem constants
#define BATCH   16
#define SEQ     1024
#define DMODEL  512
#define HEADS   8
#define DHEAD   64
#define INNER   512           // HEADS*DHEAD
#define NQKV    1536          // 3*INNER
#define MTOT    16384         // BATCH*SEQ
#define BHTOT   128           // BATCH*HEADS

// ---------------------------------------------------------------------------
// Prep: f32 -> bf16 copy, and f32 -> bf16 transpose
// ---------------------------------------------------------------------------
__global__ void cvt_bf16_kernel(const float* __restrict__ src,
                                unsigned short* __restrict__ dst, int n) {
    int i = blockIdx.x * blockDim.x + threadIdx.x;
    if (i < n) dst[i] = f2bf(src[i]);
}

__global__ void transpose_bf16_kernel(const float* __restrict__ src,  // [rows][cols]
                                      unsigned short* __restrict__ dst, // [cols][rows]
                                      int rows, int cols) {
    int i = blockIdx.x * blockDim.x + threadIdx.x;
    if (i < rows * cols) {
        int r = i / cols, c = i % cols;
        dst[c * rows + r] = f2bf(src[i]);
    }
}

// ---------------------------------------------------------------------------
// Kernel 1: QKV GEMM.  [16384,512]bf16 x [512,1536] -> scatter into
//   q,k : [b*h][n][64] bf16     v : transposed [b*h][64][n] bf16
// Block = 8 waves = 128(M) x 64(N) tile.  The 64-col weight strip
// (64KB, contiguous in wT) is staged once into LDS via async copies and
// shared by all 8 waves; each wave computes a 16x64 strip.
// ---------------------------------------------------------------------------
__global__ __launch_bounds__(256)
void qkv_gemm_kernel(const unsigned short* __restrict__ xb,    // [16384][512]
                     const unsigned short* __restrict__ wT,    // [1536][512]
                     unsigned short* __restrict__ qo,
                     unsigned short* __restrict__ ko,
                     unsigned short* __restrict__ vTo) {
    extern __shared__ unsigned short Bs[];                    // [64][512] = 64KB

    const int lane   = threadIdx.x & 31;
    const int wave   = threadIdx.x >> 5;
    const int blockM = blockIdx.x / 24;                       // 0..127
    const int tileN  = (blockIdx.x % 24) * 64;                // 0..1472
    const int tileM  = blockM * 8 + wave;                     // 16-row tile id, 0..1023
    const int mrow   = lane & 15;
    const int kh     = lane >> 4;

    // ---- stage B strip: 64KB contiguous at wT + tileN*512 ----
    {
        const char* gsrc = (const char*)(wT + (size_t)tileN * DMODEL);
        unsigned lbase = (unsigned)(size_t)(&Bs[0]);
        #pragma unroll
        for (int r = 0; r < 16; ++r) {
            unsigned ofs = (threadIdx.x + r * 256) * 16;
            async_copy16(lbase + ofs, gsrc + ofs);
        }
    }
    wait_async0();
    __syncthreads();

    v8f z = {0.f,0.f,0.f,0.f,0.f,0.f,0.f,0.f};
    v8f acc[4] = {z, z, z, z};

    const unsigned short* arow = xb + (size_t)(tileM * 16 + mrow) * DMODEL + kh * 8;
    for (int k0 = 0; k0 < DMODEL; k0 += 32) {
        Frag16 a;
        a.q[0] = *(const uint4*)(arow + k0);
        a.q[1] = *(const uint4*)(arow + k0 + 16);
        #pragma unroll
        for (int t = 0; t < 4; ++t) {
            const unsigned short* bp = Bs + (t * 16 + mrow) * DMODEL + k0 + kh * 8;
            Frag16 b;
            b.q[0] = *(const uint4*)(bp);
            b.q[1] = *(const uint4*)(bp + 16);
            acc[t] = WMMA_BF16(a.v, b.v, acc[t]);
        }
    }

    #pragma unroll
    for (int t = 0; t < 4; ++t) {
        int c = tileN + t * 16 + mrow;                // global output column
        #pragma unroll
        for (int r = 0; r < 8; ++r) {
            int m     = tileM * 16 + r + (kh ? 8 : 0);
            int bb    = m >> 10;                      // batch
            int nseq  = m & 1023;
            unsigned short bv = f2bf(acc[t][r]);
            if (c < 512) {
                int h = c >> 6, dh = c & 63;
                qo[(size_t)((bb * HEADS + h) * SEQ + nseq) * DHEAD + dh] = bv;
            } else if (c < 1024) {
                int c2 = c - 512, h = c2 >> 6, dh = c2 & 63;
                ko[(size_t)((bb * HEADS + h) * SEQ + nseq) * DHEAD + dh] = bv;
            } else {
                int c2 = c - 1024, h = c2 >> 6, dh = c2 & 63;
                vTo[(size_t)((bb * HEADS + h) * DHEAD + dh) * SEQ + nseq] = bv;
            }
        }
    }
}

// ---------------------------------------------------------------------------
// Kernel 2: attention.  One block (128 threads / 4 waves) per (b,h,qblock16).
// Phase 1: S = scale * Q K^T with diagonal -inf, f32 scores in LDS [16][1024]
// Phase 2: softmax (max / exp / sum) in LDS
// Phase 3: O = P V via WMMA, scaled by 1/rowsum, bf16 ctx out [16384][512]
// ---------------------------------------------------------------------------
__global__ __launch_bounds__(128)
void attn_kernel(const unsigned short* __restrict__ qw,
                 const unsigned short* __restrict__ kw,
                 const unsigned short* __restrict__ vTw,
                 const float* __restrict__ scale,
                 unsigned short* __restrict__ ctx) {
    extern __shared__ char smem_raw[];
    float* S      = (float*)smem_raw;                        // [16][1024]
    float* red    = (float*)(smem_raw + 16 * 1024 * 4);      // [16][8]
    float* rowmax = red + 16 * 8;                            // [16]
    float* rowinv = rowmax + 16;                             // [16]

    const int qb   = blockIdx.x & 63;
    const int bh   = blockIdx.x >> 6;                        // 0..127
    const int b    = bh >> 3, h = bh & 7;
    const int lane = threadIdx.x & 31;
    const int wave = threadIdx.x >> 5;
    const int mrow = lane & 15;
    const int kh   = lane >> 4;
    const float sc = scale[h];

    // Q fragments for this 16-row block (A: 16x64 = two 16x32 frags)
    const unsigned short* qrow = qw + (size_t)(bh * SEQ + qb * 16 + mrow) * DHEAD + kh * 8;
    Frag16 a0, a1;
    a0.q[0] = *(const uint4*)(qrow);
    a0.q[1] = *(const uint4*)(qrow + 16);
    a1.q[0] = *(const uint4*)(qrow + 32);
    a1.q[1] = *(const uint4*)(qrow + 48);

    v8f z = {0.f,0.f,0.f,0.f,0.f,0.f,0.f,0.f};

    // ---- Phase 1: scores ----
    for (int kt = wave; kt < 64; kt += 4) {
        const unsigned short* krow = kw + (size_t)(bh * SEQ + kt * 16 + mrow) * DHEAD + kh * 8;
        Frag16 b0, b1;
        b0.q[0] = *(const uint4*)(krow);
        b0.q[1] = *(const uint4*)(krow + 16);
        b1.q[0] = *(const uint4*)(krow + 32);
        b1.q[1] = *(const uint4*)(krow + 48);
        v8f accs = z;
        accs = WMMA_BF16(a0.v, b0.v, accs);
        accs = WMMA_BF16(a1.v, b1.v, accs);

        int key = kt * 16 + mrow;
        #pragma unroll
        for (int r = 0; r < 8; ++r) {
            int qr = r + (kh ? 8 : 0);
            float sv = accs[r] * sc;
            if (qb * 16 + qr == key) sv = -__builtin_inff();   // LSA diagonal mask
            S[qr * SEQ + key] = sv;
        }
    }
    __syncthreads();

    // ---- Phase 2: softmax ----
    {
        int row = threadIdx.x >> 3, seg = threadIdx.x & 7;
        const float* srow = S + row * SEQ + seg * 128;
        float m = -__builtin_inff();
        for (int j = 0; j < 128; ++j) m = fmaxf(m, srow[j]);
        red[row * 8 + seg] = m;
    }
    __syncthreads();
    if (threadIdx.x < 16) {
        float m = red[threadIdx.x * 8];
        for (int s2 = 1; s2 < 8; ++s2) m = fmaxf(m, red[threadIdx.x * 8 + s2]);
        rowmax[threadIdx.x] = m;
    }
    __syncthreads();
    {
        int row = threadIdx.x >> 3, seg = threadIdx.x & 7;
        float* srow = S + row * SEQ + seg * 128;
        float m = rowmax[row], sum = 0.f;
        for (int j = 0; j < 128; ++j) {
            float p = __expf(srow[j] - m);
            srow[j] = p;
            sum += p;
        }
        red[row * 8 + seg] = sum;
    }
    __syncthreads();
    if (threadIdx.x < 16) {
        float s = 0.f;
        for (int s2 = 0; s2 < 8; ++s2) s += red[threadIdx.x * 8 + s2];
        rowinv[threadIdx.x] = 1.0f / s;
    }
    __syncthreads();

    // ---- Phase 3: O = P V (each wave owns one 16-wide dh chunk) ----
    {
        const int dt = wave;                                  // 0..3
        v8f acc = z;
        const unsigned short* vrow = vTw + (size_t)(bh * DHEAD + dt * 16 + mrow) * SEQ + kh * 8;
        const float* prow = S + mrow * SEQ + kh * 8;
        for (int k0 = 0; k0 < SEQ; k0 += 32) {
            Frag16 a;
            #pragma unroll
            for (int e = 0; e < 8; ++e) {
                a.u[e]     = f2bf(prow[k0 + e]);
                a.u[e + 8] = f2bf(prow[k0 + 16 + e]);
            }
            Frag16 bf;
            bf.q[0] = *(const uint4*)(vrow + k0);
            bf.q[1] = *(const uint4*)(vrow + k0 + 16);
            acc = WMMA_BF16(a.v, bf.v, acc);
        }
        int dh = dt * 16 + mrow;
        #pragma unroll
        for (int r = 0; r < 8; ++r) {
            int qr = r + (kh ? 8 : 0);
            float val = acc[r] * rowinv[qr];
            int mg = b * SEQ + qb * 16 + qr;
            ctx[(size_t)mg * INNER + h * DHEAD + dh] = f2bf(val);
        }
    }
}

// ---------------------------------------------------------------------------
// Kernel 3: output projection  ctx[16384,512]bf16 x woutT + bias -> f32 out
// Same LDS-staged B-strip scheme as kernel 1.
// ---------------------------------------------------------------------------
__global__ __launch_bounds__(256)
void out_gemm_kernel(const unsigned short* __restrict__ ctxb,
                     const unsigned short* __restrict__ wT,   // [512][512]
                     const float* __restrict__ bias,
                     float* __restrict__ out) {
    extern __shared__ unsigned short Bs[];                    // [64][512] = 64KB

    const int lane   = threadIdx.x & 31;
    const int wave   = threadIdx.x >> 5;
    const int blockM = blockIdx.x >> 3;                       // 0..127
    const int tileN  = (blockIdx.x & 7) * 64;
    const int tileM  = blockM * 8 + wave;                     // 0..1023
    const int mrow   = lane & 15;
    const int kh     = lane >> 4;

    {
        const char* gsrc = (const char*)(wT + (size_t)tileN * INNER);
        unsigned lbase = (unsigned)(size_t)(&Bs[0]);
        #pragma unroll
        for (int r = 0; r < 16; ++r) {
            unsigned ofs = (threadIdx.x + r * 256) * 16;
            async_copy16(lbase + ofs, gsrc + ofs);
        }
    }
    wait_async0();
    __syncthreads();

    v8f z = {0.f,0.f,0.f,0.f,0.f,0.f,0.f,0.f};
    v8f acc[4] = {z, z, z, z};

    const unsigned short* arow = ctxb + (size_t)(tileM * 16 + mrow) * INNER + kh * 8;
    for (int k0 = 0; k0 < INNER; k0 += 32) {
        Frag16 a;
        a.q[0] = *(const uint4*)(arow + k0);
        a.q[1] = *(const uint4*)(arow + k0 + 16);
        #pragma unroll
        for (int t = 0; t < 4; ++t) {
            const unsigned short* bp = Bs + (t * 16 + mrow) * INNER + k0 + kh * 8;
            Frag16 b;
            b.q[0] = *(const uint4*)(bp);
            b.q[1] = *(const uint4*)(bp + 16);
            acc[t] = WMMA_BF16(a.v, b.v, acc[t]);
        }
    }

    #pragma unroll
    for (int t = 0; t < 4; ++t) {
        int c = tileN + t * 16 + mrow;
        float bv = bias[c];
        #pragma unroll
        for (int r = 0; r < 8; ++r) {
            int m = tileM * 16 + r + (kh ? 8 : 0);
            out[(size_t)m * DMODEL + c] = acc[t][r] + bv;
        }
    }
}

// ---------------------------------------------------------------------------
extern "C" void kernel_launch(void* const* d_in, const int* in_sizes, int n_in,
                              void* d_out, int out_size, void* d_ws, size_t ws_size,
                              hipStream_t stream) {
    const float* x     = (const float*)d_in[0];
    const float* w_qkv = (const float*)d_in[1];
    const float* w_out = (const float*)d_in[2];
    const float* b_out = (const float*)d_in[3];
    const float* scale = (const float*)d_in[4];
    float* out = (float*)d_out;

    char* ws = (char*)d_ws;
    unsigned short* xb    = (unsigned short*)ws; ws += (size_t)MTOT  * DMODEL * 2;
    unsigned short* wqkvT = (unsigned short*)ws; ws += (size_t)NQKV  * DMODEL * 2;
    unsigned short* woutT = (unsigned short*)ws; ws += (size_t)INNER * DMODEL * 2;
    unsigned short* q_ws  = (unsigned short*)ws; ws += (size_t)BHTOT * SEQ * DHEAD * 2;
    unsigned short* k_ws  = (unsigned short*)ws; ws += (size_t)BHTOT * SEQ * DHEAD * 2;
    unsigned short* vT_ws = (unsigned short*)ws; ws += (size_t)BHTOT * SEQ * DHEAD * 2;
    unsigned short* ctx   = (unsigned short*)ws; ws += (size_t)MTOT  * INNER * 2;

    // Prep: bf16 conversions / weight transposes
    {
        int n = MTOT * DMODEL;
        cvt_bf16_kernel<<<(n + 255) / 256, 256, 0, stream>>>(x, xb, n);
    }
    {
        int n = DMODEL * NQKV;
        transpose_bf16_kernel<<<(n + 255) / 256, 256, 0, stream>>>(w_qkv, wqkvT, DMODEL, NQKV);
    }
    {
        int n = INNER * DMODEL;
        transpose_bf16_kernel<<<(n + 255) / 256, 256, 0, stream>>>(w_out, woutT, INNER, DMODEL);
    }

    // QKV projection: 128 M-blocks x 24 N-strips, 64KB LDS B strip per block
    qkv_gemm_kernel<<<3072, 256, 65536, stream>>>(xb, wqkvT, q_ws, k_ws, vT_ws);

    // Attention: one block per (b,h,qblock); 64 qblocks * 128 bh = 8192 blocks
    size_t smem = (size_t)16 * SEQ * 4 + (16 * 8 + 16 + 16) * 4;
    attn_kernel<<<8192, 128, smem, stream>>>(q_ws, k_ws, vT_ws, scale, ctx);

    // Output projection: 128 M-blocks x 8 N-strips
    out_gemm_kernel<<<1024, 256, 65536, stream>>>(ctx, woutT, b_out, out);
}
